// RelationalLayerIR_37228776522175
// MI455X (gfx1250) — compile-verified, matrix operand order checked by reading, not run on hardware
//
#include <hip/hip_runtime.h>
#include <hip/hip_bf16.h>
#include <math.h>

typedef __bf16 bf16_t;
typedef __attribute__((ext_vector_type(16))) __bf16 v16bf;
typedef __attribute__((ext_vector_type(8)))  __bf16 v8bf;
typedef __attribute__((ext_vector_type(8)))  float  v8f;

// Problem constants
#define BATCH 64
#define NOBJ  64      // d*d = 8*8
#define KCH   24
#define QSTN  256
#define OUTC  28

// ---------------------------------------------------------------------------
// WMMA helpers
// ---------------------------------------------------------------------------
static __device__ __forceinline__ v8f wmma_bf16(v16bf a, v16bf b, v8f c) {
    return __builtin_amdgcn_wmma_f32_16x16x32_bf16(
        /*neg_a=*/false, a, /*neg_b=*/false, b,
        /*c_mod=*/(short)0, c, /*reuse_a=*/false, /*reuse_b=*/false);
}

// Load one lane's 16 bf16 values for a 16x16x32 fragment from row-major data.
// Per ISA layout: lanes 0-15 hold K {0..7,16..23}; lanes 16-31 hold K {8..15,24..31}.
// Caller passes (row_base + kc + koff) where koff = (lane&16)?8:0.
static __device__ __forceinline__ v16bf ld_frag(const bf16_t* __restrict__ p) {
    v8bf lo = *(const v8bf*)(p);
    v8bf hi = *(const v8bf*)(p + 16);
    v16bf r;
#pragma unroll
    for (int i = 0; i < 8; ++i) { r[i] = lo[i]; r[i + 8] = hi[i]; }
    return r;
}

// ---------------------------------------------------------------------------
// One fused GEMM layer:  out[32x256] = relu(in[32xKIN] * W^T + bias)
// Wave tiling: 8 waves = 2 (M) x 4 (N); each wave: 16 rows x 64 cols (4 tiles).
// KIN = logical K (and weight row length); SIN = LDS row stride of input.
// ---------------------------------------------------------------------------
template <int KIN, int SIN>
static __device__ __forceinline__ void gemm_acc(
    v8f (&acc)[4], const bf16_t* __restrict__ sIn, const bf16_t* __restrict__ W,
    int mrow, int koff, int ncol, int nbase)
{
#pragma unroll
    for (int kc = 0; kc < KIN; kc += 32) {
        v16bf a = ld_frag(sIn + mrow * SIN + kc + koff);
#pragma unroll
        for (int nt = 0; nt < 4; ++nt) {
            int n = nbase + (nt << 4) + ncol;
            v16bf b = ld_frag(W + n * KIN + kc + koff);
            acc[nt] = wmma_bf16(a, b, acc[nt]);
        }
    }
}

template <int KIN, int SIN, int SOUT>
static __device__ __forceinline__ void layer(
    const bf16_t* __restrict__ sIn, const bf16_t* __restrict__ W,
    const float* __restrict__ bias, bf16_t* __restrict__ sOut,
    int mtbase, int rhalf, int mrow, int koff, int ncol, int nbase)
{
    v8f acc[4] = {};
    gemm_acc<KIN, SIN>(acc, sIn, W, mrow, koff, ncol, nbase);
#pragma unroll
    for (int nt = 0; nt < 4; ++nt) {
        int n = nbase + (nt << 4) + ncol;
        float bv = bias[n];
#pragma unroll
        for (int r = 0; r < 8; ++r) {
            float v = acc[nt][r] + bv;
            v = v > 0.0f ? v : 0.0f;
            sOut[(mtbase + rhalf + r) * SOUT + n] = (bf16_t)v;
        }
    }
}

// Final (h) layer: relu then row-sum into LDS pooled accumulator.
template <int KIN, int SIN>
static __device__ __forceinline__ void layer_final(
    const bf16_t* __restrict__ sIn, const bf16_t* __restrict__ W,
    const float* __restrict__ bias, float* __restrict__ xgs,
    int mrow, int koff, int ncol, int nbase)
{
    v8f acc[4] = {};
    gemm_acc<KIN, SIN>(acc, sIn, W, mrow, koff, ncol, nbase);
#pragma unroll
    for (int nt = 0; nt < 4; ++nt) {
        int n = nbase + (nt << 4) + ncol;
        float bv = bias[n];
        float s = 0.0f;
#pragma unroll
        for (int r = 0; r < 8; ++r) {
            float v = acc[nt][r] + bv;
            s += (v > 0.0f ? v : 0.0f);
        }
        atomicAdd(&xgs[n], s);   // lanes l and l+16 share column n -> combined here
    }
}

// ---------------------------------------------------------------------------
// Kernel 1: bf16 weight/question prep + zero pooled accumulator
// ---------------------------------------------------------------------------
__global__ void prep_kernel(
    const float* __restrict__ Wg1, const float* __restrict__ Wg2,
    const float* __restrict__ Wg3, const float* __restrict__ Wg4,
    const float* __restrict__ Wh1, const float* __restrict__ qst,
    bf16_t* __restrict__ wg1p, bf16_t* __restrict__ wg2b,
    bf16_t* __restrict__ wg3b, bf16_t* __restrict__ wg4b,
    bf16_t* __restrict__ wh1b, bf16_t* __restrict__ qstb,
    float* __restrict__ xg)
{
    int i = blockIdx.x * blockDim.x + threadIdx.x;
    int stride = gridDim.x * blockDim.x;
    // Wg1: 256x52 -> padded 256x64
    for (int idx = i; idx < 256 * 64; idx += stride) {
        int o = idx >> 6, c = idx & 63;
        wg1p[idx] = (bf16_t)((c < 52) ? Wg1[o * 52 + c] : 0.0f);
    }
    for (int idx = i; idx < 256 * 256; idx += stride) wg2b[idx] = (bf16_t)Wg2[idx];
    for (int idx = i; idx < 256 * 256; idx += stride) wg3b[idx] = (bf16_t)Wg3[idx];
    for (int idx = i; idx < 256 * 256; idx += stride) wg4b[idx] = (bf16_t)Wg4[idx];
    for (int idx = i; idx < 256 * 512; idx += stride) wh1b[idx] = (bf16_t)Wh1[idx];
    for (int idx = i; idx < BATCH * QSTN; idx += stride) qstb[idx] = (bf16_t)qst[idx];
    for (int idx = i; idx < BATCH * 256; idx += stride) xg[idx] = 0.0f;
}

// ---------------------------------------------------------------------------
// Kernel 2: fused relational core. One WG = 32 pair rows (fixed b, fixed p,
// 32 consecutive q). 8192 WGs x 256 threads.
// ---------------------------------------------------------------------------
__global__ __launch_bounds__(256) void rn_pair_kernel(
    const float* __restrict__ x,          // [64,24,64] f32
    const float* __restrict__ bg1, const float* __restrict__ bg2,
    const float* __restrict__ bg3, const float* __restrict__ bg4,
    const float* __restrict__ bh1,
    const bf16_t* __restrict__ wg1, const bf16_t* __restrict__ wg2,
    const bf16_t* __restrict__ wg3, const bf16_t* __restrict__ wg4,
    const bf16_t* __restrict__ wh1, const bf16_t* __restrict__ qstb,
    float* __restrict__ xg)
{
    __shared__ bf16_t bufA[32 * 512];   // wide buffer (h0 @stride64, L2/L4 out @stride512)
    __shared__ bf16_t bufB[32 * 256];   // narrow buffer
    __shared__ float  xgs[256];

    const int tid = threadIdx.x;
    const int blk = blockIdx.x;
    const int b   = blk >> 7;            // 128 WGs per batch element
    const int rem = blk & 127;
    const int p   = rem >> 1;            // pair index p (0..63)
    const int q0  = (rem & 1) * 32;      // q range start

    xgs[tid] = 0.0f;

    // Build h0 (32 rows x 64 cols, cols 52..63 zero-padded) into bufA, stride 64.
    // Row r => pair (p, q0+r); cols 0..25 = feat(q), 26..51 = feat(p).
    const float cstep = 8.0f / 7.0f;
    for (int idx = tid; idx < 32 * 64; idx += 256) {
        int r = idx >> 6, c = idx & 63;
        float v = 0.0f;
        if (c < 52) {
            int o  = (c < 26) ? (q0 + r) : p;
            int cc = (c < 26) ? c : (c - 26);
            if (cc < 24)       v = x[((b * KCH + cc) << 6) + o];
            else if (cc == 24) v = -4.0f + (float)(o & 7)  * cstep;  // cx
            else               v = -4.0f + (float)(o >> 3) * cstep;  // cy
        }
        bufA[(r << 6) + c] = (bf16_t)v;
    }
    __syncthreads();

    const int lane   = tid & 31;
    const int wv     = tid >> 5;
    const int mt     = wv >> 2;                  // 0..1 : M subtile
    const int nq     = wv & 3;                   // 0..3 : N quarter
    const int mtbase = mt << 4;
    const int mrow   = mtbase + (lane & 15);     // A-fragment row
    const int koff   = (lane & 16) ? 8 : 0;      // K interleave offset
    const int rhalf  = (lane >> 4) << 3;         // C/D row-half offset
    const int ncol   = lane & 15;
    const int nbase  = nq << 6;

    // g-MLP chain
    layer<64, 64, 256>(bufA, wg1, bg1, bufB, mtbase, rhalf, mrow, koff, ncol, nbase);
    __syncthreads();
    layer<256, 256, 512>(bufB, wg2, bg2, bufA, mtbase, rhalf, mrow, koff, ncol, nbase);
    __syncthreads();
    // L3: K=256 valid columns read from bufA at stride 512
    layer<256, 512, 256>(bufA, wg3, bg3, bufB, mtbase, rhalf, mrow, koff, ncol, nbase);
    __syncthreads();
    layer<256, 256, 512>(bufB, wg4, bg4, bufA, mtbase, rhalf, mrow, koff, ncol, nbase);

    // Append question into bufA cols 256..511 (same for all 32 rows).
    {
        bf16_t qv = qstb[b * QSTN + tid];
#pragma unroll
        for (int r = 0; r < 32; ++r) bufA[r * 512 + 256 + tid] = qv;
    }
    __syncthreads();

    // h-layer (K=512) with fused relu + pair-sum pooling into xgs.
    layer_final<512, 512>(bufA, wh1, bh1, xgs, mrow, koff, ncol, nbase);
    __syncthreads();

    atomicAdd(&xg[b * 256 + tid], xgs[tid]);
}

// ---------------------------------------------------------------------------
// Kernel 3: f-MLP + log_softmax (tiny, FP32). One WG per batch row.
// ---------------------------------------------------------------------------
__global__ __launch_bounds__(256) void f_mlp_kernel(
    const float* __restrict__ xg,
    const float* __restrict__ Wf1, const float* __restrict__ bf1v,
    const float* __restrict__ Wf2, const float* __restrict__ bf2v,
    const float* __restrict__ Wf3, const float* __restrict__ bf3v,
    float* __restrict__ out)
{
    __shared__ float s0[256], s1[256], sl[OUTC], red[2];
    const int b = blockIdx.x, t = threadIdx.x;

    s0[t] = xg[b * 256 + t];
    __syncthreads();

    float acc = bf1v[t];
    for (int k = 0; k < 256; ++k) acc += Wf1[t * 256 + k] * s0[k];
    s1[t] = fmaxf(acc, 0.0f);
    __syncthreads();

    acc = bf2v[t];
    for (int k = 0; k < 256; ++k) acc += Wf2[t * 256 + k] * s1[k];
    s0[t] = fmaxf(acc, 0.0f);
    __syncthreads();

    if (t < OUTC) {
        acc = bf3v[t];
        for (int k = 0; k < 256; ++k) acc += Wf3[t * 256 + k] * s0[k];
        sl[t] = acc;
    }
    __syncthreads();

    if (t == 0) {
        float m = sl[0];
        for (int i = 1; i < OUTC; ++i) m = fmaxf(m, sl[i]);
        float se = 0.0f;
        for (int i = 0; i < OUTC; ++i) se += expf(sl[i] - m);
        red[0] = m;
        red[1] = logf(se);
    }
    __syncthreads();

    if (t < OUTC) out[b * OUTC + t] = sl[t] - red[0] - red[1];
}

// ---------------------------------------------------------------------------
// Host launcher
// ---------------------------------------------------------------------------
extern "C" void kernel_launch(void* const* d_in, const int* in_sizes, int n_in,
                              void* d_out, int out_size, void* d_ws, size_t ws_size,
                              hipStream_t stream) {
    (void)in_sizes; (void)n_in; (void)out_size; (void)ws_size;
    const float* x   = (const float*)d_in[0];
    const float* qst = (const float*)d_in[1];
    const float* Wg1 = (const float*)d_in[2];  const float* bg1 = (const float*)d_in[3];
    const float* Wg2 = (const float*)d_in[4];  const float* bg2 = (const float*)d_in[5];
    const float* Wg3 = (const float*)d_in[6];  const float* bg3 = (const float*)d_in[7];
    const float* Wg4 = (const float*)d_in[8];  const float* bg4 = (const float*)d_in[9];
    const float* Wh1 = (const float*)d_in[10]; const float* bh1 = (const float*)d_in[11];
    const float* Wf1 = (const float*)d_in[12]; const float* bf1 = (const float*)d_in[13];
    const float* Wf2 = (const float*)d_in[14]; const float* bf2 = (const float*)d_in[15];
    const float* Wf3 = (const float*)d_in[16]; const float* bf3 = (const float*)d_in[17];

    // Workspace layout (bf16 elements, then float region), total ~768 KB.
    bf16_t* wsb  = (bf16_t*)d_ws;
    bf16_t* wg1p = wsb + 0;        // 256*64   = 16384
    bf16_t* wg2b = wsb + 16384;    // 256*256  = 65536
    bf16_t* wg3b = wsb + 81920;
    bf16_t* wg4b = wsb + 147456;
    bf16_t* wh1b = wsb + 212992;   // 256*512  = 131072
    bf16_t* qstb = wsb + 344064;   // 64*256   = 16384
    float*  xg   = (float*)((char*)d_ws + 720896); // 64*256 f32

    prep_kernel<<<256, 256, 0, stream>>>(Wg1, Wg2, Wg3, Wg4, Wh1, qst,
                                         wg1p, wg2b, wg3b, wg4b, wh1b, qstb, xg);

    rn_pair_kernel<<<8192, 256, 0, stream>>>(x, bg1, bg2, bg3, bg4, bh1,
                                             wg1p, wg2b, wg3b, wg4b, wh1b, qstb, xg);

    f_mlp_kernel<<<64, 256, 0, stream>>>(xg, Wf1, bf1, Wf2, bf2, Wf3, bf3,
                                         (float*)d_out);
}
